// Attention_20504173871374
// MI455X (gfx1250) — matrix operand order repaired and checked by reading, not compiled, not run on hardware
//
#include <hip/hip_runtime.h>

// ---------------- types ----------------
typedef __bf16 bf16_t;
typedef bf16_t v16bf __attribute__((ext_vector_type(16)));
typedef float  v8f   __attribute__((ext_vector_type(8)));

#define DIMC 768
#define NSEQ 2048
#define BB   4
#define HH   12
#define DHH  64

#define WMMA_BF16(A, B, C) \
    __builtin_amdgcn_wmma_f32_16x16x32_bf16(false, (A), false, (B), (short)0, (C), false, false)

static __device__ __forceinline__ unsigned short f2bf(float f) {
    unsigned int u = __float_as_uint(f);
    u += 0x7FFFu + ((u >> 16) & 1u);     // round-to-nearest-even
    return (unsigned short)(u >> 16);
}

union FragAB {
    v16bf v;
    unsigned int u[8];
    unsigned short h[16];
};

// two contiguous 16B chunks -> 16 bf16 halves
static __device__ __forceinline__ void load_frag16(FragAB& f,
                                                   const unsigned short* p0,
                                                   const unsigned short* p1) {
    uint4 x = *(const uint4*)p0;
    uint4 y = *(const uint4*)p1;
    f.u[0] = x.x; f.u[1] = x.y; f.u[2] = x.z; f.u[3] = x.w;
    f.u[4] = y.x; f.u[5] = y.y; f.u[6] = y.z; f.u[7] = y.w;
}

// ---------------- kernel 1: LayerNorm (no affine) * (gamma+1) -> bf16 ----------------
__global__ void ln_kernel(const float* __restrict__ x, const float* __restrict__ gamma,
                          unsigned short* __restrict__ xn) {
    __shared__ float s_sum[256];
    __shared__ float s_sq[256];
    int row = blockIdx.x;
    int t = threadIdx.x;
    const float* xr = x + (size_t)row * DIMC;
    float v0 = xr[t], v1 = xr[t + 256], v2 = xr[t + 512];
    s_sum[t] = v0 + v1 + v2;
    s_sq[t]  = v0 * v0 + v1 * v1 + v2 * v2;
    __syncthreads();
    for (int off = 128; off > 0; off >>= 1) {
        if (t < off) { s_sum[t] += s_sum[t + off]; s_sq[t] += s_sq[t + off]; }
        __syncthreads();
    }
    float mean = s_sum[0] * (1.0f / DIMC);
    float var  = s_sq[0] * (1.0f / DIMC) - mean * mean;
    float rstd = rsqrtf(var + 1e-5f);
    unsigned short* o = xn + (size_t)row * DIMC;
    o[t]       = f2bf((v0 - mean) * rstd * (gamma[t] + 1.0f));
    o[t + 256] = f2bf((v1 - mean) * rstd * (gamma[t + 256] + 1.0f));
    o[t + 512] = f2bf((v2 - mean) * rstd * (gamma[t + 512] + 1.0f));
}

// ---------------- kernel 2: convert 4 weights to bf16  WT[n][k]=W[n][k] ----------------
__global__ void wprep_kernel(const float* __restrict__ Wq, const float* __restrict__ Wk,
                             const float* __restrict__ Wv, const float* __restrict__ Wo,
                             unsigned short* __restrict__ wt) {
    const float* W = (blockIdx.y == 0) ? Wq : (blockIdx.y == 1) ? Wk
                   : (blockIdx.y == 2) ? Wv : Wo;
    unsigned short* D = wt + (size_t)blockIdx.y * DIMC * DIMC;
    int idx = blockIdx.x * 256 + threadIdx.x;          // idx = k*DIMC + n
    int k = idx / DIMC, n = idx - k * DIMC;
    D[(size_t)n * DIMC + k] = f2bf(W[(size_t)n * DIMC + k]);
}

// ---------------- kernel 3/5: WMMA GEMM  C[M,768] = A[M,768] @ B (Bt holds B^T[n][k]) ----------------
// mode 0: store bf16 [B,H,N,DH] (q, with scale)   mode 1: store bf16 [B,H,DH,N] (vT)
// mode 2: store bf16 [B,H,N,DH] (k)               mode 3: store f32 row-major (final out)
__global__ void gemm_kernel(const unsigned short* __restrict__ A,
                            const unsigned short* __restrict__ Bt,
                            unsigned short* __restrict__ dstb,
                            float* __restrict__ dstf,
                            int mode, float scale) {
    int lane = threadIdx.x & 31;
    int wave = threadIdx.x >> 5;                        // 4 waves, stacked in M
    int hi = lane >> 4, lm = lane & 15;
    int m0 = blockIdx.x * 64 + wave * 16;
    int n0 = blockIdx.y * 64;

    v8f zero = {0.f, 0.f, 0.f, 0.f, 0.f, 0.f, 0.f, 0.f};
    v8f acc[4];
#pragma unroll
    for (int i = 0; i < 4; i++) acc[i] = zero;

    const unsigned short* arow = A + (size_t)(m0 + lm) * DIMC;
    for (int k = 0; k < DIMC; k += 32) {
        FragAB a;
        load_frag16(a, arow + k + 8 * hi, arow + k + 16 + 8 * hi);
#pragma unroll
        for (int nf = 0; nf < 4; nf++) {
            const unsigned short* brow = Bt + (size_t)(n0 + nf * 16 + lm) * DIMC + k + 8 * hi;
            FragAB b;
            load_frag16(b, brow, brow + 16);
            acc[nf] = WMMA_BF16(a.v, b.v, acc[nf]);
        }
    }

    if (mode == 3) {
#pragma unroll
        for (int nf = 0; nf < 4; nf++) {
            int col = n0 + nf * 16 + lm;
#pragma unroll
            for (int r = 0; r < 8; r++) {
                int grow = m0 + r + 8 * hi;
                dstf[(size_t)grow * DIMC + col] = acc[nf][r];
            }
        }
    } else {
#pragma unroll
        for (int nf = 0; nf < 4; nf++) {
            int col = n0 + nf * 16 + lm;
            int h = col >> 6, d = col & 63;
#pragma unroll
            for (int r = 0; r < 8; r++) {
                int grow = m0 + r + 8 * hi;
                int bb = grow >> 11, nrow = grow & (NSEQ - 1);
                size_t idx;
                if (mode == 1)
                    idx = (((size_t)(bb * HH + h)) * DHH + d) * NSEQ + nrow;     // V^T
                else
                    idx = (((size_t)(bb * HH + h)) * NSEQ + nrow) * DHH + d;     // Q / K
                dstb[idx] = f2bf(acc[nf][r] * scale);
            }
        }
    }
}

// ---------------- kernel 4: flash attention core ----------------
// q: [BH][N][64]  kT: [BH][N][64]  v: [BH][64][N]  ->  ao: [B*N][768] bf16
__global__ void attn_kernel(const unsigned short* __restrict__ q,
                            const unsigned short* __restrict__ kT,
                            const unsigned short* __restrict__ v,
                            unsigned short* __restrict__ ao) {
    __shared__ unsigned short plds[4][16][32];          // per-wave P staging (4 KB)
    int lane = threadIdx.x & 31, wave = threadIdx.x >> 5;
    int hi = lane >> 4, lm = lane & 15;
    int bh = blockIdx.y;
    int h = bh % HH, b = bh / HH;
    int q0 = blockIdx.x * 64 + wave * 16;               // 16 Q rows per wave

    const unsigned short* qp = q  + (size_t)bh * NSEQ * DHH;
    const unsigned short* kp = kT + (size_t)bh * DHH * NSEQ;
    const unsigned short* vp = v  + (size_t)bh * NSEQ * DHH;

    // Q A-fragments, K-dim chunks d=[0,32) and [32,64)
    const unsigned short* qrow = qp + (size_t)(q0 + lm) * DHH;
    FragAB aq0, aq1;
    load_frag16(aq0, qrow + 8 * hi,      qrow + 16 + 8 * hi);
    load_frag16(aq1, qrow + 32 + 8 * hi, qrow + 48 + 8 * hi);

    v8f zero = {0.f, 0.f, 0.f, 0.f, 0.f, 0.f, 0.f, 0.f};
    v8f o[4];
#pragma unroll
    for (int i = 0; i < 4; i++) o[i] = zero;
    v8f mrow, lrow = zero;
#pragma unroll
    for (int r = 0; r < 8; r++) mrow[r] = -1e30f;

    for (int jb = 0; jb < NSEQ; jb += 32) {
        // S = Q K^T for columns [jb, jb+32): left/right 16-wide subtiles
        v8f sl = zero, sr = zero;
        {
            const unsigned short* kr0 = kp + (size_t)(jb + lm) * DHH + 8 * hi;
            FragAB bl, br;
            load_frag16(bl, kr0, kr0 + 16);
            load_frag16(br, kr0 + 16 * DHH, kr0 + 16 * DHH + 16);
            sl = WMMA_BF16(aq0.v, bl.v, sl);
            sr = WMMA_BF16(aq0.v, br.v, sr);
            const unsigned short* kr1 = kr0 + 32;
            load_frag16(bl, kr1, kr1 + 16);
            load_frag16(br, kr1 + 16 * DHH, kr1 + 16 * DHH + 16);
            sl = WMMA_BF16(aq1.v, bl.v, sl);
            sr = WMMA_BF16(aq1.v, br.v, sr);
        }
        // online softmax: row m = r + 8*hi lives across the 16 lanes of the half-wave
        v8f tmax, mnew, alpha, pl, pr, rsum;
#pragma unroll
        for (int r = 0; r < 8; r++) tmax[r] = fmaxf(sl[r], sr[r]);
#pragma unroll
        for (int mask = 1; mask <= 8; mask <<= 1)
#pragma unroll
            for (int r = 0; r < 8; r++) tmax[r] = fmaxf(tmax[r], __shfl_xor(tmax[r], mask, 32));
#pragma unroll
        for (int r = 0; r < 8; r++) {
            mnew[r]  = fmaxf(mrow[r], tmax[r]);
            alpha[r] = __expf(mrow[r] - mnew[r]);
            pl[r]    = __expf(sl[r] - mnew[r]);
            pr[r]    = __expf(sr[r] - mnew[r]);
            rsum[r]  = pl[r] + pr[r];
        }
#pragma unroll
        for (int mask = 1; mask <= 8; mask <<= 1)
#pragma unroll
            for (int r = 0; r < 8; r++) rsum[r] += __shfl_xor(rsum[r], mask, 32);
#pragma unroll
        for (int r = 0; r < 8; r++) {
            lrow[r] = lrow[r] * alpha[r] + rsum[r];
            mrow[r] = mnew[r];
        }
#pragma unroll
        for (int nf = 0; nf < 4; nf++)
#pragma unroll
            for (int r = 0; r < 8; r++) o[nf][r] *= alpha[r];

        // C-layout P -> A-layout bf16 via per-wave LDS transpose
#pragma unroll
        for (int r = 0; r < 8; r++) {
            plds[wave][r + 8 * hi][lm]      = f2bf(pl[r]);
            plds[wave][r + 8 * hi][16 + lm] = f2bf(pr[r]);
        }
        asm volatile("s_wait_dscnt 0x0" ::: "memory");
        FragAB pa;
        const unsigned short* prow = &plds[wave][lm][0];
        load_frag16(pa, prow + 8 * hi, prow + 16 + 8 * hi);

        // O += P(16x32) x V(32x64)
#pragma unroll
        for (int nf = 0; nf < 4; nf++) {
            const unsigned short* vr = vp + (size_t)(nf * 16 + lm) * NSEQ + jb + 8 * hi;
            FragAB bv;
            load_frag16(bv, vr, vr + 16);
            o[nf] = WMMA_BF16(pa.v, bv.v, o[nf]);
        }
    }

    // epilogue: O /= l, store bf16 to [B*N][768] at head column block
    v8f inv;
#pragma unroll
    for (int r = 0; r < 8; r++) inv[r] = 1.0f / lrow[r];
#pragma unroll
    for (int nf = 0; nf < 4; nf++)
#pragma unroll
        for (int r = 0; r < 8; r++) {
            int grow = b * NSEQ + q0 + r + 8 * hi;
            ao[(size_t)grow * DIMC + h * DHH + nf * 16 + lm] = f2bf(o[nf][r] * inv[r]);
        }
}

// ---------------- launch ----------------
extern "C" void kernel_launch(void* const* d_in, const int* in_sizes, int n_in,
                              void* d_out, int out_size, void* d_ws, size_t ws_size,
                              hipStream_t stream) {
    const float* x     = (const float*)d_in[0];
    const float* gamma = (const float*)d_in[1];
    const float* Wq    = (const float*)d_in[2];
    const float* Wk    = (const float*)d_in[3];
    const float* Wv    = (const float*)d_in[4];
    const float* Wo    = (const float*)d_in[5];
    float* out = (float*)d_out;

    const size_t ROWS = (size_t)BB * NSEQ;            // 8192
    const size_t TOK  = ROWS * DIMC;                  // 6,291,456 elems
    const size_t WSZ  = (size_t)DIMC * DIMC;          // 589,824 elems

    unsigned short* ws  = (unsigned short*)d_ws;
    unsigned short* xn  = ws;                         // [8192][768]
    unsigned short* wt  = xn + TOK;                   // 4 x [768][768]  (WqT,WkT,WvT,WoT)
    unsigned short* WqT = wt;
    unsigned short* WkT = wt + WSZ;
    unsigned short* WvT = wt + 2 * WSZ;
    unsigned short* WoT = wt + 3 * WSZ;
    unsigned short* qb  = wt + 4 * WSZ;               // [BH][N][64]
    unsigned short* kb  = qb + TOK;                   // [BH][N][64]
    unsigned short* vb  = kb + TOK;                   // [BH][64][N]
    unsigned short* aob = vb + TOK;                   // [8192][768]

    ln_kernel<<<dim3((unsigned)ROWS), 256, 0, stream>>>(x, gamma, xn);
    wprep_kernel<<<dim3(WSZ / 256, 4), 256, 0, stream>>>(Wq, Wk, Wv, Wo, wt);

    dim3 ggrid(ROWS / 64, DIMC / 64);                 // 128 x 12
    gemm_kernel<<<ggrid, 128, 0, stream>>>(xn, WqT, qb, nullptr, 0, 0.125f);  // q * DH^-0.5
    gemm_kernel<<<ggrid, 128, 0, stream>>>(xn, WkT, kb, nullptr, 2, 1.0f);    // k
    gemm_kernel<<<ggrid, 128, 0, stream>>>(xn, WvT, vb, nullptr, 1, 1.0f);    // v transposed

    attn_kernel<<<dim3(NSEQ / 64, BB * HH), 128, 0, stream>>>(qb, kb, vb, aob);

    gemm_kernel<<<ggrid, 128, 0, stream>>>(aob, WoT, nullptr, out, 3, 1.0f);  // final proj, f32
}
